// SurfaceToVolumeModel_35330400977028
// MI455X (gfx1250) — compile-verified, matrix operand order checked by reading, not run on hardware
//
#include <hip/hip_runtime.h>
#include <hip/hip_bf16.h>
#include <math.h>

#define BATCH 2
#define S_PTS 4096
#define T_PTS 8192
#define KNN_E 20
#define KNN_T 8

typedef __attribute__((ext_vector_type(16))) __bf16 v16bf;
typedef __attribute__((ext_vector_type(8)))  __bf16 v8bf;
typedef __attribute__((ext_vector_type(8)))  float  v8f;

static_assert(S_PTS % 256 == 0, "knn kernel assumes S multiple of 256");

// Load one lane's 16x16x32 bf16 fragment from a packed row-major operand.
// Per CDNA5 ISA 7.12.2: lane's K elements are two contiguous runs of 8:
//   e in [0,8):  K = 8*half + e
//   e in [8,16): K = 16 + 8*half + (e-8)
// -> exactly two b128 loads per fragment.
static __device__ __forceinline__ v16bf load_frag(const __bf16* __restrict__ row,
                                                  int kbase, int half) {
  v8bf lo = *(const v8bf*)(row + kbase + 8 * half);
  v8bf hi = *(const v8bf*)(row + kbase + 16 + 8 * half);
  v16bf f;
#pragma unroll
  for (int e = 0; e < 8; ++e) { f[e] = lo[e]; f[8 + e] = hi[e]; }
  return f;
}

static __device__ __forceinline__ void atomicMaxF(float* addr, float val) {
  int* ai = (int*)addr;
  int old = __float_as_int(*addr);
  while (__int_as_float(old) < val) {
    int assumed = old;
    old = atomicCAS(ai, assumed, __float_as_int(val));
    if (old == assumed) break;
  }
}

// ---------------------------------------------------------------- utilities
__global__ void k_fill(float* __restrict__ p, float v, int n) {
  int i = blockIdx.x * blockDim.x + threadIdx.x;
  if (i < n) p[i] = v;
}

__global__ void k_transpose_surface(const float* __restrict__ surf,
                                    float* __restrict__ x0) {
  int i = blockIdx.x * blockDim.x + threadIdx.x;  // over B*S*6
  if (i >= BATCH * S_PTS * 6) return;
  int c = i % 6;
  int s = (i / 6) % S_PTS;
  int b = i / (6 * S_PTS);
  x0[(b * 6 + c) * S_PTS + s] = surf[i];
}

__global__ void k_xx(const float* __restrict__ x, int bstride, int d,
                     float* __restrict__ xx) {
  int m = blockIdx.x * blockDim.x + threadIdx.x;
  int b = blockIdx.y;
  if (m >= S_PTS) return;
  const float* xb = x + b * bstride;
  float s = 0.f;
  for (int c = 0; c < d; ++c) {
    float v = xb[c * S_PTS + m];
    s += v * v;
  }
  xx[b * S_PTS + m] = s;
}

// ------------------------------------------------------------------- kNN
__global__ void k_knn(const float* __restrict__ x, int bstride, int d,
                      const float* __restrict__ xx,
                      const float* __restrict__ query,
                      int kn, int* __restrict__ idx_out) {
  __shared__ float sc[S_PTS];
  __shared__ float ctr[128];
  __shared__ float rv[256];
  __shared__ int   ri[256];
  const int tid = threadIdx.x;
  const int n = blockIdx.x, b = blockIdx.y;
  const int rows = gridDim.x;
  const float* xb = x + b * bstride;

  if (query) {
    for (int c = tid; c < d; c += 256) ctr[c] = query[(b * rows + n) * 3 + c];
  } else {
    for (int c = tid; c < d; c += 256) ctr[c] = xb[c * S_PTS + n];
  }
  __syncthreads();

  float acc[S_PTS / 256];
#pragma unroll
  for (int j = 0; j < S_PTS / 256; ++j) acc[j] = 0.f;
  for (int c = 0; c < d; ++c) {
    float qc = ctr[c];
    const float* row = xb + c * S_PTS;
#pragma unroll
    for (int j = 0; j < S_PTS / 256; ++j) acc[j] += qc * row[tid + j * 256];
  }
#pragma unroll
  for (int j = 0; j < S_PTS / 256; ++j)
    sc[tid + j * 256] = xx[b * S_PTS + tid + j * 256] - 2.f * acc[j];
  __syncthreads();

  for (int it = 0; it < kn; ++it) {
    float bv = __builtin_inff();
    int bi = 0x7fffffff;
#pragma unroll
    for (int j = 0; j < S_PTS / 256; ++j) {
      int m = tid + j * 256;
      float v = sc[m];
      if (v < bv || (v == bv && m < bi)) { bv = v; bi = m; }
    }
    rv[tid] = bv; ri[tid] = bi;
    __syncthreads();
    for (int off = 128; off > 0; off >>= 1) {
      if (tid < off) {
        if (rv[tid + off] < rv[tid] ||
            (rv[tid + off] == rv[tid] && ri[tid + off] < ri[tid])) {
          rv[tid] = rv[tid + off];
          ri[tid] = ri[tid + off];
        }
      }
      __syncthreads();
    }
    if (tid == 0) {
      idx_out[(b * rows + n) * kn + it] = ri[0];
      sc[ri[0]] = __builtin_inff();
    }
    __syncthreads();
  }
}

// ------------------------------------------------------------ pack kernels
__global__ void k_pack_w(const float* __restrict__ w, int ktot, int kp, int n,
                         __bf16* __restrict__ wbf) {
  int i = blockIdx.x * blockDim.x + threadIdx.x;
  if (i >= n * kp) return;
  int o = i / kp, c = i % kp;
  wbf[i] = (__bf16)((c < ktot) ? w[o * ktot + c] : 0.f);
}

__global__ void k_pack_ef(const float* __restrict__ x, int bstride, int din,
                          int kp, const int* __restrict__ idx,
                          __bf16* __restrict__ ef) {
  const int ctot = 2 * din;
  size_t i = (size_t)blockIdx.x * blockDim.x + threadIdx.x;
  size_t total = (size_t)BATCH * S_PTS * KNN_E * kp;
  if (i >= total) return;
  int c = (int)(i % kp);
  int m = (int)((i / kp) % (S_PTS * KNN_E));
  int b = (int)(i / ((size_t)kp * S_PTS * KNN_E));
  const float* xb = x + b * bstride;
  int n = m / KNN_E, k = m % KNN_E;
  float v = 0.f;
  if (c < din) {
    int nb = idx[(b * S_PTS + n) * KNN_E + k];
    v = xb[c * S_PTS + nb] - xb[c * S_PTS + n];
  } else if (c < ctot) {
    v = xb[(c - din) * S_PTS + n];
  }
  ef[i] = (__bf16)v;
}

__global__ void k_pack_cat(const float* __restrict__ cat,
                           __bf16* __restrict__ catbf) {
  size_t i = (size_t)blockIdx.x * blockDim.x + threadIdx.x;
  size_t total = (size_t)BATCH * S_PTS * 448;
  if (i >= total) return;
  int c = (int)(i % 448);
  int m = (int)((i / 448) % S_PTS);
  int b = (int)(i / ((size_t)448 * S_PTS));
  catbf[i] = (__bf16)cat[((size_t)b * 448 + c) * S_PTS + m];
}

// ----------------------------------------------------- group-norm finalize
__global__ void k_gn_finalize(const float* __restrict__ gsum,
                              const float* __restrict__ gsq,
                              float* __restrict__ gmean,
                              float* __restrict__ grstd, int groups,
                              float cnt) {
  int i = threadIdx.x;
  if (i < BATCH * groups) {
    float mu = gsum[i] / cnt;
    float var = gsq[i] / cnt - mu * mu;
    gmean[i] = mu;
    grstd[i] = rsqrtf(var + 1e-5f);
  }
}

// ------------------------------------------------------- WMMA macro-tile
// Each wave owns an (MB*16) x (NB*16) macro-tile: MB A-fragments and NB
// B-fragments per K-step feed MB*NB back-to-back WMMAs (1.5 loads/WMMA at
// MB=2,NB=4 vs 4 loads/WMMA unblocked).
template <int MB, int NB>
struct MacroAcc { v8f t[MB][NB]; };

template <int MB, int NB>
static __device__ __forceinline__ void gemm_core(
    const __bf16* __restrict__ abase, size_t arowstride_rows_kp_m0,
    const __bf16* __restrict__ bbase, int kp, int l16, int half,
    MacroAcc<MB, NB>& acc) {
  const __bf16* arow[MB];
  const __bf16* brow[NB];
#pragma unroll
  for (int i = 0; i < MB; ++i)
    arow[i] = abase + (size_t)(arowstride_rows_kp_m0 + i * 16 + l16) * kp;
#pragma unroll
  for (int j = 0; j < NB; ++j)
    brow[j] = bbase + (size_t)(j * 16 + l16) * kp;

  for (int kbase = 0; kbase < kp; kbase += 32) {
    v16bf a[MB], bm[NB];
#pragma unroll
    for (int i = 0; i < MB; ++i) a[i] = load_frag(arow[i], kbase, half);
#pragma unroll
    for (int j = 0; j < NB; ++j) bm[j] = load_frag(brow[j], kbase, half);
#pragma unroll
    for (int i = 0; i < MB; ++i)
#pragma unroll
      for (int j = 0; j < NB; ++j)
        acc.t[i][j] = __builtin_amdgcn_wmma_f32_16x16x32_bf16(
            false, a[i], false, bm[j], (short)0, acc.t[i][j], false, false);
  }
}

// ---------------------------------------------- edge-conv GEMM (WMMA bf16)
template <bool STATS, int MB, int NB>
__global__ void k_edge_gemm(const __bf16* __restrict__ ef, int kp,
                            const __bf16* __restrict__ wbf, int cout,
                            float* __restrict__ gsum, float* __restrict__ gsq,
                            const float* __restrict__ gmean,
                            const float* __restrict__ grstd,
                            const float* __restrict__ gamma,
                            const float* __restrict__ beta, int groups,
                            float* __restrict__ out, int ostride) {
  const int tid = threadIdx.x;
  const int lane = tid & 31, wave = tid >> 5;
  const int half = lane >> 4, l16 = lane & 15;
  const int b = blockIdx.z;
  const int mtile0 = (blockIdx.x * 4 + wave) * 16 * MB;
  const int ntile0 = blockIdx.y * 16 * NB;

  MacroAcc<MB, NB> acc = {};
  gemm_core<MB, NB>(ef, (size_t)b * S_PTS * KNN_E + mtile0,
                    wbf + (size_t)ntile0 * kp, kp, l16, half, acc);

  if constexpr (STATS) {
    __shared__ float lsum[16 * NB], lsq[16 * NB];
    if (tid < 16 * NB) { lsum[tid] = 0.f; lsq[tid] = 0.f; }
    __syncthreads();
#pragma unroll
    for (int j = 0; j < NB; ++j) {
      float s = 0.f, q = 0.f;
#pragma unroll
      for (int i = 0; i < MB; ++i)
#pragma unroll
        for (int r = 0; r < 8; ++r) {
          float v = acc.t[i][j][r];
          s += v; q += v * v;
        }
      atomicAdd(&lsum[j * 16 + l16], s);
      atomicAdd(&lsq[j * 16 + l16], q);
    }
    __syncthreads();
    if (tid < 16 * NB) {
      int g = (ntile0 + tid) / (cout / groups);
      atomicAdd(&gsum[b * groups + g], lsum[tid]);
      atomicAdd(&gsq[b * groups + g], lsq[tid]);
    }
  } else {
#pragma unroll
    for (int j = 0; j < NB; ++j) {
      const int oB = ntile0 + j * 16 + l16;
      const int g = oB / (cout / groups);
      const float mu = gmean[b * groups + g];
      const float rs = grstd[b * groups + g];
      const float ga = gamma[oB], be = beta[oB];
#pragma unroll
      for (int i = 0; i < MB; ++i)
#pragma unroll
        for (int r = 0; r < 8; ++r) {
          int mrow = mtile0 + i * 16 + r + 8 * half;
          float y = (acc.t[i][j][r] - mu) * rs * ga + be;
          y = (y >= 0.f) ? y : 0.2f * y;
          atomicMaxF(&out[b * ostride + oB * S_PTS + mrow / KNN_E], y);
        }
    }
  }
}

// ---------------------------------- pp / ga GEMM over packed cat, WMMA bf16
// MODE 0: stats, MODE 1: store pf (B,256,S) f32, MODE 2: global max/sum.
template <int MODE, int MB, int NB>
__global__ void k_plain_gemm(const __bf16* __restrict__ A, int kp, int cout,
                             const __bf16* __restrict__ wbf,
                             float* __restrict__ gsum, float* __restrict__ gsq,
                             const float* __restrict__ gmean,
                             const float* __restrict__ grstd,
                             const float* __restrict__ gamma,
                             const float* __restrict__ beta, int groups,
                             float* __restrict__ out, int ostride,
                             float* __restrict__ gmax,
                             float* __restrict__ gsumf) {
  const int tid = threadIdx.x;
  const int lane = tid & 31, wave = tid >> 5;
  const int half = lane >> 4, l16 = lane & 15;
  const int b = blockIdx.z;
  const int mtile0 = (blockIdx.x * 4 + wave) * 16 * MB;
  const int ntile0 = blockIdx.y * 16 * NB;

  MacroAcc<MB, NB> acc = {};
  gemm_core<MB, NB>(A, (size_t)b * S_PTS + mtile0,
                    wbf + (size_t)ntile0 * kp, kp, l16, half, acc);

  if constexpr (MODE == 0) {
    __shared__ float lsum[16 * NB], lsq[16 * NB];
    if (tid < 16 * NB) { lsum[tid] = 0.f; lsq[tid] = 0.f; }
    __syncthreads();
#pragma unroll
    for (int j = 0; j < NB; ++j) {
      float s = 0.f, q = 0.f;
#pragma unroll
      for (int i = 0; i < MB; ++i)
#pragma unroll
        for (int r = 0; r < 8; ++r) {
          float v = acc.t[i][j][r];
          s += v; q += v * v;
        }
      atomicAdd(&lsum[j * 16 + l16], s);
      atomicAdd(&lsq[j * 16 + l16], q);
    }
    __syncthreads();
    if (tid < 16 * NB) {
      int g = (ntile0 + tid) / (cout / groups);
      atomicAdd(&gsum[b * groups + g], lsum[tid]);
      atomicAdd(&gsq[b * groups + g], lsq[tid]);
    }
  } else {
#pragma unroll
    for (int j = 0; j < NB; ++j) {
      const int oB = ntile0 + j * 16 + l16;
      const int g = oB / (cout / groups);
      const float mu = gmean[b * groups + g];
      const float rs = grstd[b * groups + g];
      const float ga = gamma[oB], be = beta[oB];
#pragma unroll
      for (int i = 0; i < MB; ++i)
#pragma unroll
        for (int r = 0; r < 8; ++r) {
          int mrow = mtile0 + i * 16 + r + 8 * half;
          float y = (acc.t[i][j][r] - mu) * rs * ga + be;
          y = (y >= 0.f) ? y : 0.2f * y;
          if constexpr (MODE == 1) {
            out[b * ostride + oB * S_PTS + mrow] = y;
          } else {
            atomicMaxF(&gmax[b * cout + oB], y);
            atomicAdd(&gsumf[b * cout + oB], y);
          }
        }
    }
  }
}

__global__ void k_gfeat(const float* __restrict__ gmax,
                        const float* __restrict__ gsumf,
                        float* __restrict__ gf) {
  int i = blockIdx.x * blockDim.x + threadIdx.x;
  if (i >= BATCH * 256) return;
  int b = i / 256, c = i % 256;
  gf[b * 512 + c] = gmax[i];
  gf[b * 512 + 256 + c] = gsumf[i] / (float)S_PTS;
}

// node rows in packed bf16 (B*T, 800): [tmpl(3)|local(256)|gfeat(512)|pad(29)]
__global__ void k_node(const float* __restrict__ tmpl,
                       const float* __restrict__ pf,
                       const int* __restrict__ nn, const float* __restrict__ gf,
                       __bf16* __restrict__ node) {
  int row = blockIdx.x;  // 0 .. B*T-1
  int b = row / T_PTS;
  for (int c = threadIdx.x; c < 800; c += blockDim.x) {
    float v = 0.f;
    if (c < 3) {
      v = tmpl[row * 3 + c];
    } else if (c < 259) {
      const float* pfb = pf + (size_t)(b * 256 + (c - 3)) * S_PTS;
      float s = 0.f;
#pragma unroll
      for (int j = 0; j < KNN_T; ++j) s += pfb[nn[row * KNN_T + j]];
      v = s * (1.f / KNN_T);
    } else if (c < 771) {
      v = gf[b * 512 + (c - 259)];
    }
    node[(size_t)row * 800 + c] = (__bf16)v;
  }
}

// -------------------------------------------- decoder MLP GEMM (WMMA bf16)
template <int ACT, bool BF16OUT, int MB, int NB>  // ACT: 0 none, 1 relu, 2 sig
__global__ void k_mlp(const __bf16* __restrict__ A, int kp,
                      const __bf16* __restrict__ wbf,
                      const float* __restrict__ bias, int nout,
                      float* __restrict__ outf, __bf16* __restrict__ outbf) {
  const int tid = threadIdx.x;
  const int lane = tid & 31, wave = tid >> 5;
  const int half = lane >> 4, l16 = lane & 15;
  const int mtile0 = (blockIdx.x * 4 + wave) * 16 * MB;
  const int ntile0 = blockIdx.y * 16 * NB;

  MacroAcc<MB, NB> acc = {};
  gemm_core<MB, NB>(A, (size_t)mtile0, wbf + (size_t)ntile0 * kp, kp, l16,
                    half, acc);

#pragma unroll
  for (int j = 0; j < NB; ++j) {
    const int oB = ntile0 + j * 16 + l16;
    const float bo = (oB < nout) ? bias[oB] : 0.f;
#pragma unroll
    for (int i = 0; i < MB; ++i)
#pragma unroll
      for (int r = 0; r < 8; ++r) {
        float v = acc.t[i][j][r] + bo;
        if (ACT == 1) v = fmaxf(v, 0.f);
        if (ACT == 2) v = 1.f / (1.f + __expf(-v));
        int row = mtile0 + i * 16 + r + 8 * half;
        if (BF16OUT) {
          outbf[(size_t)row * nout + oB] = (__bf16)v;
        } else if (oB < nout) {
          outf[(size_t)row * nout + oB] = v;
        }
      }
  }
}

// ---------------------------------------------------------------- launcher
extern "C" void kernel_launch(void* const* d_in, const int* in_sizes, int n_in,
                              void* d_out, int out_size, void* d_ws,
                              size_t ws_size, hipStream_t stream) {
  (void)in_sizes; (void)n_in; (void)out_size; (void)ws_size;
  const float* surface = (const float*)d_in[0];
  const float* tmpl    = (const float*)d_in[1];
  const float* ec1_w = (const float*)d_in[2];
  const float* ec1_g = (const float*)d_in[3];
  const float* ec1_b = (const float*)d_in[4];
  const float* ec2_w = (const float*)d_in[5];
  const float* ec2_g = (const float*)d_in[6];
  const float* ec2_b = (const float*)d_in[7];
  const float* ec3_w = (const float*)d_in[8];
  const float* ec3_g = (const float*)d_in[9];
  const float* ec3_b = (const float*)d_in[10];
  const float* pp_w = (const float*)d_in[11];
  const float* pp_g = (const float*)d_in[12];
  const float* pp_b = (const float*)d_in[13];
  const float* ga_w = (const float*)d_in[14];
  const float* ga_g = (const float*)d_in[15];
  const float* ga_b = (const float*)d_in[16];
  const float* d1_w = (const float*)d_in[17];
  const float* d1_b = (const float*)d_in[18];
  const float* d2_w = (const float*)d_in[19];
  const float* d2_b = (const float*)d_in[20];
  const float* d3_w = (const float*)d_in[21];
  const float* d3_b = (const float*)d_in[22];
  const float* m1_w = (const float*)d_in[23];
  const float* m1_b = (const float*)d_in[24];
  const float* m2_w = (const float*)d_in[25];
  const float* m2_b = (const float*)d_in[26];
  const float* m3_w = (const float*)d_in[27];
  const float* m3_b = (const float*)d_in[28];

  // ---- workspace bump allocator (256B aligned)
  size_t off = 0;
  auto alloc = [&](size_t bytes) {
    void* p = (char*)d_ws + off;
    off += (bytes + 255) & ~(size_t)255;
    return p;
  };
  float*  x0    = (float*)alloc(sizeof(float) * BATCH * 6 * S_PTS);
  float*  cat   = (float*)alloc(sizeof(float) * BATCH * 448 * S_PTS);
  __bf16* catbf = (__bf16*)alloc(sizeof(__bf16) * (size_t)BATCH * S_PTS * 448);
  float*  xx    = (float*)alloc(sizeof(float) * BATCH * S_PTS);
  int*    idxE  = (int*)alloc(sizeof(int) * BATCH * S_PTS * KNN_E);
  int*    nnT   = (int*)alloc(sizeof(int) * BATCH * T_PTS * KNN_T);
  float*  gsum  = (float*)alloc(sizeof(float) * 64);
  float*  gsq   = gsum + 32;
  float*  gmean = (float*)alloc(sizeof(float) * 64);
  float*  grstd = gmean + 32;
  float*  pf    = (float*)alloc(sizeof(float) * BATCH * 256 * S_PTS);
  float*  gmax  = (float*)alloc(sizeof(float) * BATCH * 256);
  float*  gsumf = (float*)alloc(sizeof(float) * BATCH * 256);
  float*  gf    = (float*)alloc(sizeof(float) * BATCH * 512);
  __bf16* efbf  = (__bf16*)alloc(sizeof(__bf16) * (size_t)BATCH * S_PTS * KNN_E * 256);
  __bf16* wbf   = (__bf16*)alloc(sizeof(__bf16) * 256 * 800);
  __bf16* nodeb = (__bf16*)alloc(sizeof(__bf16) * (size_t)BATCH * T_PTS * 800);
  __bf16* h1    = (__bf16*)alloc(sizeof(__bf16) * (size_t)BATCH * T_PTS * 256);
  __bf16* h2    = (__bf16*)alloc(sizeof(__bf16) * (size_t)BATCH * T_PTS * 256);
  __bf16* hm1   = (__bf16*)alloc(sizeof(__bf16) * (size_t)BATCH * T_PTS * 128);
  __bf16* hm2   = (__bf16*)alloc(sizeof(__bf16) * (size_t)BATCH * T_PTS * 64);

  const float NEG_INF = -__builtin_inff();

  {
    int n = BATCH * S_PTS * 6;
    k_transpose_surface<<<(n + 255) / 256, 256, 0, stream>>>(surface, x0);
    int nc = BATCH * 448 * S_PTS;
    k_fill<<<(nc + 255) / 256, 256, 0, stream>>>(cat, NEG_INF, nc);
  }

  // ---- one edge-conv stage (macro-tile 32x64 per wave)
  auto edge_stage = [&](const float* xs, int bstride, int din, int cout,
                        int groups, const float* w, const float* g,
                        const float* bt, float* out_slice) {
    const int kp = (2 * din + 31) & ~31;
    k_xx<<<dim3(S_PTS / 256, BATCH), 256, 0, stream>>>(xs, bstride, din, xx);
    k_knn<<<dim3(S_PTS, BATCH), 256, 0, stream>>>(xs, bstride, din, xx,
                                                  nullptr, KNN_E, idxE);
    size_t nef = (size_t)BATCH * S_PTS * KNN_E * kp;
    k_pack_ef<<<(int)((nef + 255) / 256), 256, 0, stream>>>(xs, bstride, din,
                                                            kp, idxE, efbf);
    int nw = cout * kp;
    k_pack_w<<<(nw + 255) / 256, 256, 0, stream>>>(w, 2 * din, kp, cout, wbf);
    k_fill<<<1, 64, 0, stream>>>(gsum, 0.f, 64);
    dim3 gs(S_PTS * KNN_E / 128, cout / 64, BATCH);  // MB=2, NB=4
    k_edge_gemm<true, 2, 4><<<gs, 128, 0, stream>>>(
        efbf, kp, wbf, cout, gsum, gsq, nullptr, nullptr, nullptr, nullptr,
        groups, nullptr, 0);
    k_gn_finalize<<<1, 32, 0, stream>>>(
        gsum, gsq, gmean, grstd, groups,
        (float)((cout / groups) * S_PTS * KNN_E));
    k_edge_gemm<false, 2, 4><<<gs, 128, 0, stream>>>(
        efbf, kp, wbf, cout, nullptr, nullptr, gmean, grstd, g, bt, groups,
        out_slice, 448 * S_PTS);
  };

  // cat channel layout: x1 @ 0..63, x2 @ 64..191, x3 @ 192..447
  edge_stage(x0, 6 * S_PTS, 6, 64, 8, ec1_w, ec1_g, ec1_b, cat + 0 * S_PTS);
  edge_stage(cat + 0 * S_PTS, 448 * S_PTS, 64, 128, 8, ec2_w, ec2_g, ec2_b,
             cat + 64 * S_PTS);
  edge_stage(cat + 64 * S_PTS, 448 * S_PTS, 128, 256, 16, ec3_w, ec3_g, ec3_b,
             cat + 192 * S_PTS);

  // ---- pack cat for the 1x1 conv branches
  {
    size_t n = (size_t)BATCH * S_PTS * 448;
    k_pack_cat<<<(int)((n + 255) / 256), 256, 0, stream>>>(cat, catbf);
  }

  // ---- pp branch -> pf (B,256,S)
  {
    k_pack_w<<<(256 * 448 + 255) / 256, 256, 0, stream>>>(pp_w, 448, 448, 256,
                                                          wbf);
    k_fill<<<1, 64, 0, stream>>>(gsum, 0.f, 64);
    dim3 gp(S_PTS / 128, 256 / 64, BATCH);  // MB=2, NB=4
    k_plain_gemm<0, 2, 4><<<gp, 128, 0, stream>>>(
        catbf, 448, 256, wbf, gsum, gsq, nullptr, nullptr, nullptr, nullptr,
        16, nullptr, 0, nullptr, nullptr);
    k_gn_finalize<<<1, 32, 0, stream>>>(gsum, gsq, gmean, grstd, 16,
                                        (float)(16 * S_PTS));
    k_plain_gemm<1, 2, 4><<<gp, 128, 0, stream>>>(
        catbf, 448, 256, wbf, nullptr, nullptr, gmean, grstd, pp_g, pp_b, 16,
        pf, 256 * S_PTS, nullptr, nullptr);
  }

  // ---- ga branch -> global_feat (B,512)
  {
    k_pack_w<<<(256 * 448 + 255) / 256, 256, 0, stream>>>(ga_w, 448, 448, 256,
                                                          wbf);
    k_fill<<<1, 64, 0, stream>>>(gsum, 0.f, 64);
    int n = BATCH * 256;
    k_fill<<<(n + 255) / 256, 256, 0, stream>>>(gmax, NEG_INF, n);
    k_fill<<<(n + 255) / 256, 256, 0, stream>>>(gsumf, 0.f, n);
    dim3 gp(S_PTS / 128, 256 / 64, BATCH);
    k_plain_gemm<0, 2, 4><<<gp, 128, 0, stream>>>(
        catbf, 448, 256, wbf, gsum, gsq, nullptr, nullptr, nullptr, nullptr,
        16, nullptr, 0, nullptr, nullptr);
    k_gn_finalize<<<1, 32, 0, stream>>>(gsum, gsq, gmean, grstd, 16,
                                        (float)(16 * S_PTS));
    k_plain_gemm<2, 2, 4><<<gp, 128, 0, stream>>>(
        catbf, 448, 256, wbf, nullptr, nullptr, gmean, grstd, ga_g, ga_b, 16,
        nullptr, 0, gmax, gsumf);
    k_gfeat<<<(n + 255) / 256, 256, 0, stream>>>(gmax, gsumf, gf);
  }

  // ---- template -> surface kNN (xyz only), node features (packed bf16)
  k_xx<<<dim3(S_PTS / 256, BATCH), 256, 0, stream>>>(x0, 6 * S_PTS, 3, xx);
  k_knn<<<dim3(T_PTS, BATCH), 256, 0, stream>>>(x0, 6 * S_PTS, 3, xx, tmpl,
                                                KNN_T, nnT);
  k_node<<<BATCH * T_PTS, 256, 0, stream>>>(tmpl, pf, nnT, gf, nodeb);

  // ---- decoder MLPs (bf16 chained activations), macro-tiled
  const int RT = BATCH * T_PTS;  // 16384 rows -> RT/128 M-blocks at MB=2
  float* disp = (float*)d_out;
  float* mat  = (float*)d_out + (size_t)RT * 3;

  k_pack_w<<<(256 * 800 + 255) / 256, 256, 0, stream>>>(d1_w, 771, 800, 256,
                                                        wbf);
  k_mlp<1, true, 2, 4><<<dim3(RT / 128, 4), 128, 0, stream>>>(
      nodeb, 800, wbf, d1_b, 256, nullptr, h1);
  k_pack_w<<<(256 * 256 + 255) / 256, 256, 0, stream>>>(d2_w, 256, 256, 256,
                                                        wbf);
  k_mlp<1, true, 2, 4><<<dim3(RT / 128, 4), 128, 0, stream>>>(
      h1, 256, wbf, d2_b, 256, nullptr, h2);
  k_pack_w<<<(16 * 256 + 255) / 256, 256, 0, stream>>>(d3_w, 256, 256, 16,
                                                       wbf);
  k_mlp<0, false, 2, 1><<<dim3(RT / 128, 1), 128, 0, stream>>>(
      h2, 256, wbf, d3_b, 3, disp, nullptr);
  k_pack_w<<<(128 * 800 + 255) / 256, 256, 0, stream>>>(m1_w, 771, 800, 128,
                                                        wbf);
  k_mlp<1, true, 2, 4><<<dim3(RT / 128, 2), 128, 0, stream>>>(
      nodeb, 800, wbf, m1_b, 128, nullptr, hm1);
  k_pack_w<<<(64 * 128 + 255) / 256, 256, 0, stream>>>(m2_w, 128, 128, 64,
                                                       wbf);
  k_mlp<1, true, 2, 4><<<dim3(RT / 128, 1), 128, 0, stream>>>(
      hm1, 128, wbf, m2_b, 64, nullptr, hm2);
  k_pack_w<<<(16 * 64 + 255) / 256, 256, 0, stream>>>(m3_w, 64, 64, 16, wbf);
  k_mlp<2, false, 2, 1><<<dim3(RT / 128, 1), 128, 0, stream>>>(
      hm2, 64, wbf, m3_b, 1, mat, nullptr);
}
